// CausalSelfAttention_15977278341342
// MI455X (gfx1250) — matrix operand order repaired and checked by reading, not compiled, not run on hardware
//
#include <hip/hip_runtime.h>

// ---------------------------------------------------------------------------
// CausalSelfAttention (GQA + QK RMSNorm + RoPE) for MI455X / gfx1250.
// All matmuls on v_wmma_f32_16x16x32_bf16 with f32 accumulation.
// One-time f32->bf16 conversion of x and weights; GEMM B-tiles staged in LDS
// via gfx1250 async-to-LDS loads (ASYNCcnt), double buffered, K-step 64.
// B=2, T=2048, C=2048, NH=32, NKV=8, HD=64.
// ---------------------------------------------------------------------------

#define B_   2
#define T_   2048
#define C_   2048
#define NH_  32
#define NKV_ 8
#define HD_  64

typedef __attribute__((ext_vector_type(16))) __bf16       bf16x16;
typedef __attribute__((ext_vector_type(8)))  float        f32x8;
typedef __attribute__((ext_vector_type(4)))  float        f32x4;
typedef __attribute__((ext_vector_type(4)))  unsigned int u32x4;

union BFrag {
  bf16x16 v;
  u32x4   u[2];
};

__device__ __forceinline__ f32x8 wmma_bf16(bf16x16 a, bf16x16 b, f32x8 c) {
  // D = A(16x32 bf16) * B(32x16 bf16) + C(16x16 f32)
  return __builtin_amdgcn_wmma_f32_16x16x32_bf16(false, a, false, b,
                                                 (short)0, c, false, false);
}

// ---------------------------------------------------------------------------
// Elementwise f32 -> bf16 (8 elements / thread, b128 in, b128 out)
// ---------------------------------------------------------------------------
__global__ __launch_bounds__(256)
void cvt_f32_to_bf16(const float* __restrict__ src, __bf16* __restrict__ dst,
                     int n) {
  const int i = (blockIdx.x * 256 + threadIdx.x) * 8;
  if (i >= n) return;
  const f32x4 a = *(const f32x4*)(src + i);
  const f32x4 b = *(const f32x4*)(src + i + 4);
  union { __bf16 h[8]; u32x4 u; } o;
#pragma unroll
  for (int j = 0; j < 4; j++) {
    o.h[j]     = (__bf16)a[j];
    o.h[4 + j] = (__bf16)b[j];
  }
  *(u32x4*)(dst + i) = o.u;
}

// ---------------------------------------------------------------------------
// bf16 GEMM: C[M,N] = A[M,K] * W[N,K]^T (both bf16 row-major), C f32.
// Block = 8 waves -> 128x64 tile, K-step 64. The 64x64 B tile per K-step is
// fetched once per block with 2x global_load_async_to_lds_b128 per thread
// (8KB), double-buffered, shared by all 8 waves (8x B-traffic reduction).
// LDS B layout: column j (0..63) holds its 64 K-values contiguously (128B).
// Rotated loop: prologue issue, steady-state issue+wait(2), epilogue wait(0).
// ---------------------------------------------------------------------------
__global__ __launch_bounds__(256)
void gemm_bf16(const __bf16* __restrict__ A, const __bf16* __restrict__ W,
               float* __restrict__ Cout, int M, int N, int K) {
  __shared__ __bf16 btile[2][64 * 64];  // 2 x 8KB

  const int tid  = threadIdx.x;
  const int lane = tid & 31;
  const int wv   = tid >> 5;  // wave 0..7
  const int ntiles = N >> 6;
  const int mb = blockIdx.x / ntiles;
  const int nt = blockIdx.x - mb * ntiles;
  const int m0 = mb * 128 + wv * 16;
  const int n0 = nt << 6;

  // A addressing (16-bit A-frag layout: lane<16 -> K {0..7,16..23}, else +8)
  const int rowA = m0 + (lane & 15);
  const int kA   = (lane < 16) ? 0 : 8;
  const __bf16* pa = A + (size_t)rowA * K + kA;
  const int cs = lane & 15;
  const int kB = (lane < 16) ? 0 : 8;

  // B staging: thread t covers (col = t/8, 16B chunk q = t%8) and (col+32, q)
  const int scol = tid >> 3;
  const int sq   = tid & 7;
  const __bf16* gb  = W + (size_t)(n0 + scol) * K + sq * 8;
  const __bf16* gb2 = gb + (size_t)32 * K;
  const unsigned lds0 =
      (unsigned)(size_t)(&btile[0][0]) + scol * 128 + sq * 16;
  const unsigned lds1 =
      (unsigned)(size_t)(&btile[1][0]) + scol * 128 + sq * 16;

  f32x8 acc[4];
#pragma unroll
  for (int j = 0; j < 4; j++) acc[j] = {};

  auto issue2 = [&](unsigned lds, int kk) {
    const unsigned long long ga0 = (unsigned long long)(size_t)(gb + kk);
    const unsigned long long ga1 = (unsigned long long)(size_t)(gb2 + kk);
    asm volatile("global_load_async_to_lds_b128 %0, %1, off" ::"v"(lds),
                 "v"(ga0)
                 : "memory");
    asm volatile("global_load_async_to_lds_b128 %0, %1, off" ::"v"(lds + 4096u),
                 "v"(ga1)
                 : "memory");
  };

  auto compute = [&](const __bf16* bt, int kk) {
    BFrag a0, a1;
    a0.u[0] = *(const u32x4*)(pa + kk);
    a0.u[1] = *(const u32x4*)(pa + kk + 16);
    a1.u[0] = *(const u32x4*)(pa + kk + 32);
    a1.u[1] = *(const u32x4*)(pa + kk + 48);
#pragma unroll
    for (int j = 0; j < 4; j++) {
      const __bf16* pb = bt + (j * 16 + cs) * 64 + kB;
      BFrag b0, b1;
      b0.u[0] = *(const u32x4*)(pb);
      b0.u[1] = *(const u32x4*)(pb + 16);
      b1.u[0] = *(const u32x4*)(pb + 32);
      b1.u[1] = *(const u32x4*)(pb + 48);
      acc[j] = wmma_bf16(a0.v, b0.v, acc[j]);
      acc[j] = wmma_bf16(a1.v, b1.v, acc[j]);
    }
  };

  issue2(lds0, 0);  // prologue: stage K-block 0 into buffer 0
  int kk = 0;
  for (; kk < K - 64; kk += 64) {
    const int cur = (kk >> 6) & 1;
    issue2(cur ? lds0 : lds1, kk + 64);  // stage next into other buffer
    // async loads complete in order: <=2 outstanding => current tile done
    asm volatile("s_wait_asynccnt 0x2" ::: "memory");
    __syncthreads();  // all waves' staging of `cur` visible
    compute(&btile[cur][0], kk);
    __syncthreads();  // everyone done reading `cur` before it is re-staged
  }
  asm volatile("s_wait_asynccnt 0x0" ::: "memory");
  __syncthreads();
  compute(&btile[(kk >> 6) & 1][0], kk);

  const int rowoff = (lane < 16) ? 0 : 8;
#pragma unroll
  for (int j = 0; j < 4; j++) {
#pragma unroll
    for (int r = 0; r < 8; r++) {
      Cout[(size_t)(m0 + rowoff + r) * N + (n0 + j * 16 + cs)] = acc[j][r];
    }
  }
}

// ---------------------------------------------------------------------------
// Prep: RMSNorm(q,k per head) + RoPE + fold 1/sqrt(HD) into q; emit bf16
// q[B,NH,T,64], k[B,NKV,T,64], vT[B,NKV,64,T] (transposed for PV WMMA).
// One wave per (b, t, head) row; lane handles elements j and j+32.
// ---------------------------------------------------------------------------
__global__ __launch_bounds__(256)
void prep_norm_rope(const float* __restrict__ qf, const float* __restrict__ kf,
                    const float* __restrict__ vf,
                    const float* __restrict__ qw, const float* __restrict__ kw,
                    __bf16* __restrict__ qh, __bf16* __restrict__ kh,
                    __bf16* __restrict__ vt) {
  const int lane = threadIdx.x & 31;
  int wid = blockIdx.x * (blockDim.x >> 5) + (threadIdx.x >> 5);
  const int QROWS = B_ * T_ * NH_;
  const int KROWS = B_ * T_ * NKV_;
  const float LN1E4_OVER_32 = 0.2878231366242557f;  // ln(10000)/32

  if (wid < QROWS) {
    const int h = wid % NH_;
    const int t = (wid / NH_) % T_;
    const int b = wid / (NH_ * T_);
    const float* p = qf + (size_t)(b * T_ + t) * C_ + h * HD_;
    float x1 = p[lane], x2 = p[lane + 32];
    float ss = x1 * x1 + x2 * x2;
#pragma unroll
    for (int off = 16; off >= 1; off >>= 1) ss += __shfl_xor(ss, off);
    const float inv = rsqrtf(ss * (1.0f / 64.0f) + 1e-6f);
    const float n1 = x1 * inv * qw[lane];
    const float n2 = x2 * inv * qw[lane + 32];
    const float ang = (float)t * __expf(-(float)lane * LN1E4_OVER_32);
    float s, c;
    __sincosf(ang, &s, &c);
    const float o1 = (n1 * c - n2 * s) * 0.125f;  // fold 1/sqrt(64)
    const float o2 = (n2 * c + n1 * s) * 0.125f;
    __bf16* dst = qh + ((size_t)(b * NH_ + h) * T_ + t) * HD_;
    dst[lane]      = (__bf16)o1;
    dst[lane + 32] = (__bf16)o2;
  } else if (wid < QROWS + KROWS) {
    wid -= QROWS;
    const int g = wid % NKV_;
    const int t = (wid / NKV_) % T_;
    const int b = wid / (NKV_ * T_);
    const float* p = kf + (size_t)(b * T_ + t) * (NKV_ * HD_) + g * HD_;
    float x1 = p[lane], x2 = p[lane + 32];
    float ss = x1 * x1 + x2 * x2;
#pragma unroll
    for (int off = 16; off >= 1; off >>= 1) ss += __shfl_xor(ss, off);
    const float inv = rsqrtf(ss * (1.0f / 64.0f) + 1e-6f);
    const float n1 = x1 * inv * kw[lane];
    const float n2 = x2 * inv * kw[lane + 32];
    const float ang = (float)t * __expf(-(float)lane * LN1E4_OVER_32);
    float s, c;
    __sincosf(ang, &s, &c);
    const float o1 = n1 * c - n2 * s;
    const float o2 = n2 * c + n1 * s;
    __bf16* dst = kh + ((size_t)(b * NKV_ + g) * T_ + t) * HD_;
    dst[lane]      = (__bf16)o1;
    dst[lane + 32] = (__bf16)o2;
  } else {
    wid -= QROWS + KROWS;
    const int g = wid % NKV_;
    const int t = (wid / NKV_) % T_;
    const int b = wid / (NKV_ * T_);
    const float* p = vf + (size_t)(b * T_ + t) * (NKV_ * HD_) + g * HD_;
    const float x1 = p[lane], x2 = p[lane + 32];
    __bf16* dst = vt + (size_t)(b * NKV_ + g) * HD_ * T_;
    dst[(size_t)lane * T_ + t]        = (__bf16)x1;
    dst[(size_t)(lane + 32) * T_ + t] = (__bf16)x2;
  }
}

// ---------------------------------------------------------------------------
// Flash attention: one wave per (b, h, 16-row Q tile). Online softmax over
// 32-key blocks. P goes through LDS to convert C-layout -> A-layout.
// Output written as bf16 into [B*T, C] layout for the O-projection GEMM.
// ---------------------------------------------------------------------------
__global__ __launch_bounds__(256)
void flash_attn(const __bf16* __restrict__ qh, const __bf16* __restrict__ kh,
                const __bf16* __restrict__ vt, __bf16* __restrict__ attout) {
  __shared__ __bf16 ptile[8][16 * 32];  // 1 KB per wave

  const int lane = threadIdx.x & 31;
  const int widx = threadIdx.x >> 5;
  const int wg   = blockIdx.x * 8 + widx;
  const int qt   = wg & 127;  // T/16 = 128 tiles
  const int h    = (wg >> 7) & 31;
  const int b    = wg >> 12;
  const int g    = h >> 2;  // GQA: 4 Q heads per KV head
  const int qbase = qt << 4;

  const __bf16* Q  = qh + ((size_t)(b * NH_ + h) * T_ + qbase) * HD_;
  const __bf16* Kp = kh + (size_t)(b * NKV_ + g) * T_ * HD_;
  const __bf16* Vp = vt + (size_t)(b * NKV_ + g) * HD_ * T_;

  const int cs = lane & 15;
  const int kA = (lane < 16) ? 0 : 8;
  const int kB = (lane < 16) ? 0 : 8;
  const int rb = (lane < 16) ? 0 : 8;

  BFrag qa0, qa1;
  {
    const __bf16* p = Q + cs * HD_;
    qa0.u[0] = *(const u32x4*)(p + kA);
    qa0.u[1] = *(const u32x4*)(p + kA + 16);
    qa1.u[0] = *(const u32x4*)(p + 32 + kA);
    qa1.u[1] = *(const u32x4*)(p + 32 + kA + 16);
  }

  f32x8 o0 = {}, o1 = {}, o2 = {}, o3 = {};
  f32x8 mr, lr;
#pragma unroll
  for (int r = 0; r < 8; r++) { mr[r] = -1e30f; lr[r] = 0.f; }

  const int nblk = (qbase + 47) >> 5;  // ceil((qbase+16)/32)
  for (int kb = 0; kb < nblk; kb++) {
    const int kbase = kb << 5;

    BFrag k00, k01, k10, k11;
    {
      const __bf16* p0 = Kp + (size_t)(kbase + cs) * HD_ + kB;
      const __bf16* p1 = Kp + (size_t)(kbase + 16 + cs) * HD_ + kB;
      k00.u[0] = *(const u32x4*)(p0);      k00.u[1] = *(const u32x4*)(p0 + 16);
      k01.u[0] = *(const u32x4*)(p0 + 32); k01.u[1] = *(const u32x4*)(p0 + 48);
      k10.u[0] = *(const u32x4*)(p1);      k10.u[1] = *(const u32x4*)(p1 + 16);
      k11.u[0] = *(const u32x4*)(p1 + 32); k11.u[1] = *(const u32x4*)(p1 + 48);
    }
    f32x8 S0 = {}, S1 = {};
    S0 = wmma_bf16(qa0.v, k00.v, S0);
    S0 = wmma_bf16(qa1.v, k01.v, S0);
    S1 = wmma_bf16(qa0.v, k10.v, S1);
    S1 = wmma_bf16(qa1.v, k11.v, S1);

    if (kb == nblk - 1) {  // only the last block can be causally partial
#pragma unroll
      for (int r = 0; r < 8; r++) {
        const int rowq = qbase + rb + r;
        if (kbase + cs      > rowq) S0[r] = -1e30f;
        if (kbase + 16 + cs > rowq) S1[r] = -1e30f;
      }
    }

    f32x8 mb;
#pragma unroll
    for (int r = 0; r < 8; r++) mb[r] = fmaxf(S0[r], S1[r]);
#pragma unroll
    for (int off = 1; off < 16; off <<= 1)
#pragma unroll
      for (int r = 0; r < 8; r++) mb[r] = fmaxf(mb[r], __shfl_xor(mb[r], off));

    f32x8 alpha;
#pragma unroll
    for (int r = 0; r < 8; r++) {
      const float mn = fmaxf(mr[r], mb[r]);
      alpha[r] = __expf(mr[r] - mn);
      mr[r] = mn;
    }
    f32x8 rs;
#pragma unroll
    for (int r = 0; r < 8; r++) {
      S0[r] = __expf(S0[r] - mr[r]);
      S1[r] = __expf(S1[r] - mr[r]);
      rs[r] = S0[r] + S1[r];
    }
#pragma unroll
    for (int off = 1; off < 16; off <<= 1)
#pragma unroll
      for (int r = 0; r < 8; r++) rs[r] += __shfl_xor(rs[r], off);
#pragma unroll
    for (int r = 0; r < 8; r++) {
      lr[r] = lr[r] * alpha[r] + rs[r];
      o0[r] *= alpha[r]; o1[r] *= alpha[r];
      o2[r] *= alpha[r]; o3[r] *= alpha[r];
    }

    // P: C-layout -> LDS -> A-layout (per-wave tile; DS ops are in-order)
    __bf16* tile = ptile[widx];
#pragma unroll
    for (int r = 0; r < 8; r++) {
      tile[(rb + r) * 32 + cs]      = (__bf16)S0[r];
      tile[(rb + r) * 32 + 16 + cs] = (__bf16)S1[r];
    }
    asm volatile("s_wait_dscnt 0x0" ::: "memory");
    BFrag pf;
    pf.u[0] = *(const u32x4*)(tile + cs * 32 + kA);
    pf.u[1] = *(const u32x4*)(tile + cs * 32 + kA + 16);

    BFrag v0, v1, v2, v3;
    {
      const __bf16* pv = Vp + kbase + kB;
      v0.u[0] = *(const u32x4*)(pv + (size_t)(cs)*T_);
      v0.u[1] = *(const u32x4*)(pv + (size_t)(cs)*T_ + 16);
      v1.u[0] = *(const u32x4*)(pv + (size_t)(16 + cs) * T_);
      v1.u[1] = *(const u32x4*)(pv + (size_t)(16 + cs) * T_ + 16);
      v2.u[0] = *(const u32x4*)(pv + (size_t)(32 + cs) * T_);
      v2.u[1] = *(const u32x4*)(pv + (size_t)(32 + cs) * T_ + 16);
      v3.u[0] = *(const u32x4*)(pv + (size_t)(48 + cs) * T_);
      v3.u[1] = *(const u32x4*)(pv + (size_t)(48 + cs) * T_ + 16);
    }
    o0 = wmma_bf16(pf.v, v0.v, o0);
    o1 = wmma_bf16(pf.v, v1.v, o1);
    o2 = wmma_bf16(pf.v, v2.v, o2);
    o3 = wmma_bf16(pf.v, v3.v, o3);
  }

#pragma unroll
  for (int r = 0; r < 8; r++) {
    const float li = 1.0f / lr[r];
    const size_t base =
        (size_t)(b * T_ + qbase + rb + r) * C_ + h * HD_ + cs;
    attout[base + 0]  = (__bf16)(o0[r] * li);
    attout[base + 16] = (__bf16)(o1[r] * li);
    attout[base + 32] = (__bf16)(o2[r] * li);
    attout[base + 48] = (__bf16)(o3[r] * li);
  }
}

// ---------------------------------------------------------------------------
// Launch: cvt(x, Wq, Wk, Wv, Wo) -> QKV GEMMs -> prep -> flash -> O-proj.
// Workspace layout (bytes):
//   xh   bf16 [4096,2048] @ 0          (16.8 MB) -- reused as bf16 attout
//   wqh  bf16 [2048,2048] @ 16777216   ( 8.4 MB)
//   wkh  bf16 [ 512,2048] @ 25165824   ( 2.1 MB)
//   wvh  bf16 [ 512,2048] @ 27262976   ( 2.1 MB)
//   woh  bf16 [2048,2048] @ 29360128   ( 8.4 MB)
//   qf   f32  [4096,2048] @ 37748736   (33.5 MB)
//   kf   f32  [4096, 512] @ 71303168   ( 8.4 MB)
//   vf   f32  [4096, 512] @ 79691776   ( 8.4 MB)
//   qh   bf16 [B,NH,T,64] @ 88080384   (16.8 MB)
//   kh   bf16 [B,KV,T,64] @ 104857600  ( 4.2 MB)
//   vT   bf16 [B,KV,64,T] @ 109051904  ( 4.2 MB)   total ~108 MB
// ---------------------------------------------------------------------------
extern "C" void kernel_launch(void* const* d_in, const int* in_sizes, int n_in,
                              void* d_out, int out_size, void* d_ws,
                              size_t ws_size, hipStream_t stream) {
  const float* x  = (const float*)d_in[0];
  const float* Wq = (const float*)d_in[1];
  const float* Wk = (const float*)d_in[2];
  const float* Wv = (const float*)d_in[3];
  const float* Wo = (const float*)d_in[4];
  const float* qw = (const float*)d_in[5];
  const float* kw = (const float*)d_in[6];
  float* out = (float*)d_out;

  char* ws = (char*)d_ws;
  const size_t M = (size_t)B_ * T_;  // 4096
  __bf16* xh  = (__bf16*)(ws + 0);
  __bf16* wqh = (__bf16*)(ws + 16777216);
  __bf16* wkh = (__bf16*)(ws + 25165824);
  __bf16* wvh = (__bf16*)(ws + 27262976);
  __bf16* woh = (__bf16*)(ws + 29360128);
  float*  qf  = (float*)(ws + 37748736);
  float*  kf  = (float*)(ws + 71303168);
  float*  vf  = (float*)(ws + 79691776);
  __bf16* qh  = (__bf16*)(ws + 88080384);
  __bf16* kh  = (__bf16*)(ws + 104857600);
  __bf16* vt  = (__bf16*)(ws + 109051904);
  __bf16* attout = xh;  // xh dead after QKV projections; reuse

  const dim3 blk(256);
  // one-time bf16 conversions
  cvt_f32_to_bf16<<<4096, blk, 0, stream>>>(x, xh, 8388608);
  cvt_f32_to_bf16<<<2048, blk, 0, stream>>>(Wq, wqh, 4194304);
  cvt_f32_to_bf16<<<512,  blk, 0, stream>>>(Wk, wkh, 1048576);
  cvt_f32_to_bf16<<<512,  blk, 0, stream>>>(Wv, wvh, 1048576);
  cvt_f32_to_bf16<<<2048, blk, 0, stream>>>(Wo, woh, 4194304);
  // projections: block = 128x64 tile -> (4096/128)*(N/64) blocks
  gemm_bf16<<<1024, blk, 0, stream>>>(xh, wqh, qf, (int)M, C_, C_);
  gemm_bf16<<<256,  blk, 0, stream>>>(xh, wkh, kf, (int)M, 512, C_);
  gemm_bf16<<<256,  blk, 0, stream>>>(xh, wvh, vf, (int)M, 512, C_);
  // prep: 131072 + 32768 + 32768 = 196608 waves -> 24576 blocks
  prep_norm_rope<<<24576, blk, 0, stream>>>(qf, kf, vf, qw, kw, qh, kh, vt);
  // flash: 2*32*128 = 8192 waves -> 1024 blocks
  flash_attn<<<1024, blk, 0, stream>>>(qh, kh, vt, attout);
  // output projection
  gemm_bf16<<<1024, blk, 0, stream>>>(attout, woh, out, (int)M, C_, C_);
}
